// MultiHeadedAttention_78889959293646
// MI455X (gfx1250) — matrix operand order repaired and checked by reading, not compiled, not run on hardware
//
#include <hip/hip_runtime.h>

typedef __attribute__((ext_vector_type(16))) _Float16 v16h;
typedef __attribute__((ext_vector_type(8)))  float    v8f;

#define SEQ    176      // 11 * 16
#define HNUM   8
#define HDIM   64
#define DIN    128
#define DMODEL 512
#define NEGV   (-9e15f)

// LDS element offsets (_Float16 units), padded pitches for bank-conflict-free b128 access
#define XP 136          // 128 + 8
#define WP 136
#define QP 72           // 64 + 8
#define VP 200          // 192 + 8
#define SKEYS 192       // keys padded 176 -> 192 for k=32 WMMA steps

#define OFF_X   0                   // 176*136 = 23936   (phase 1)
#define OFF_W   23936               // 64*136  = 8704    (phase 1)
#define OFF_P   0                   // 8*16*200 = 25600  (phase 2, overlaps X/W)
#define OFF_Q   32640               // 176*72 = 12672
#define OFF_K   45312               // 176*72 = 12672
#define OFF_VT  57984               // 64*200 = 12800
#define HALF_TOTAL 70784
#define SMEM_BYTES (HALF_TOTAL*2 + 8*16*4)   // + per-wave row-sum transpose buffer

union AB16 { uint4 u[2]; v16h h; };
union H8   { uint4 u;    _Float16 h[8]; };
union H4   { uint2 u;    _Float16 h[4]; };

__global__ __launch_bounds__(256, 1)
void mha_temporal_kernel(const float* __restrict__ x,
                         const float* __restrict__ Wq, const float* __restrict__ bq,
                         const float* __restrict__ Wk, const float* __restrict__ bk,
                         const float* __restrict__ Wv, const float* __restrict__ bv,
                         float* __restrict__ out)
{
    __shared__ __align__(16) unsigned char smem_raw[SMEM_BYTES];
    _Float16* sh   = (_Float16*)smem_raw;
    float*    sums = (float*)(smem_raw + HALF_TOTAL * 2);

    const int tid   = threadIdx.x;
    const int lane  = tid & 31;
    const int w     = tid >> 5;         // wave id, 0..7
    const int ln    = lane & 15;        // N / row index within 16-group
    const int hiK8  = (lane >> 4) << 3; // 0 or 8  (A-operand K base / D-row offset)
    const int hiK16 = (lane >> 4) << 4; // 0 or 16 (unused)

    const int b = blockIdx.x >> 3;
    const int h = blockIdx.x & 7;

    const float* xb = x + (size_t)b * SEQ * DIN;

    // ---------------- phase 0: stage x -> f16 LDS, zero Vt key-pad ----------------
    for (int i = tid; i < (SEQ * DIN) / 4; i += 256) {
        int s  = i >> 5;
        int d4 = (i & 31) << 2;
        float4 v = *(const float4*)(xb + s * DIN + d4);
        H4 p;
        p.h[0] = (_Float16)v.x; p.h[1] = (_Float16)v.y;
        p.h[2] = (_Float16)v.z; p.h[3] = (_Float16)v.w;
        *(uint2*)&sh[OFF_X + s * XP + d4] = p.u;
    }
    for (int i = tid; i < 64 * 16; i += 256) {
        sh[OFF_VT + (i >> 4) * VP + SEQ + (i & 15)] = (_Float16)0.f;
    }

    // weight slice loader: W[k][h*64+n] -> Wt_lds[n][k] (transposed, f16)
    auto loadW = [&](const float* W) {
        for (int i = tid; i < DIN * HDIM; i += 256) {
            int n = i & 63, k = i >> 6;
            sh[OFF_W + n * WP + k] = (_Float16)W[k * DMODEL + h * HDIM + n];
        }
    };

    // projection: C[176x64] = X[176x128] * Wslice[128x64] + bias  (f16 out)
    auto project = [&](int dstOff, const float* bias, bool relu, bool transposed) {
        for (int t = w; t < 44; t += 8) {
            int mt = t % 11, nt = t / 11;
            float bval = bias[h * HDIM + nt * 16 + ln];
            v8f acc = {};
            const int arow = mt * 16 + ln;
            const int brow = nt * 16 + ln;
#pragma unroll
            for (int ks = 0; ks < 4; ++ks) {
                AB16 A, Bm;
                const int ak = ks * 32 + hiK8;
                A.u[0]  = *(const uint4*)&sh[OFF_X + arow * XP + ak];
                A.u[1]  = *(const uint4*)&sh[OFF_X + arow * XP + ak + 16];
                const int bk2 = ks * 32 + hiK8;
                Bm.u[0] = *(const uint4*)&sh[OFF_W + brow * WP + bk2];
                Bm.u[1] = *(const uint4*)&sh[OFF_W + brow * WP + bk2 + 16];
                acc = __builtin_amdgcn_wmma_f32_16x16x32_f16(false, A.h, false, Bm.h,
                                                             (short)0, acc, false, false);
            }
            if (!transposed) {
#pragma unroll
                for (int j = 0; j < 8; ++j) {
                    float v = acc[j] + bval;
                    sh[dstOff + (mt * 16 + j + hiK8) * QP + nt * 16 + ln] = (_Float16)v;
                }
            } else {
                H8 p;
#pragma unroll
                for (int j = 0; j < 8; ++j) {
                    float v = acc[j] + bval;
                    if (relu) v = fmaxf(v, 0.f);
                    p.h[j] = (_Float16)v;
                }
                // Vt[d][m], 8 consecutive m per lane -> single b128 store
                *(uint4*)&sh[OFF_VT + (nt * 16 + ln) * VP + mt * 16 + hiK8] = p.u;
            }
        }
    };

    loadW(Wq);
    __syncthreads();
    project(OFF_Q, bq, false, false);
    __syncthreads();
    loadW(Wk);
    __syncthreads();
    project(OFF_K, bk, false, false);
    __syncthreads();
    loadW(Wv);
    __syncthreads();
    project(OFF_VT, bv, true, true);
    __syncthreads();

    // ---------------- phase 2: attention (per-wave mtiles, P buffers overlap X/W) --
    const int pbase = OFF_P + w * 16 * VP;
    for (int i = lane; i < 16 * 16; i += 32) {
        sh[pbase + (i >> 4) * VP + SEQ + (i & 15)] = (_Float16)0.f;  // P pad cols
    }

    for (int mt = w; mt < 11; mt += 8) {
        const int q0 = mt * 16;
        float sc[11][8];
        float mx[8];
#pragma unroll
        for (int j = 0; j < 8; ++j) mx[j] = -3.4e38f;

#pragma unroll
        for (int nt = 0; nt < 11; ++nt) {
            v8f acc = {};
            const int qrow = q0 + ln;
            const int krow = nt * 16 + ln;
#pragma unroll
            for (int ks = 0; ks < 2; ++ks) {
                AB16 A, Bm;
                const int ak = ks * 32 + hiK8;
                A.u[0]  = *(const uint4*)&sh[OFF_Q + qrow * QP + ak];
                A.u[1]  = *(const uint4*)&sh[OFF_Q + qrow * QP + ak + 16];
                const int bk2 = ks * 32 + hiK8;
                Bm.u[0] = *(const uint4*)&sh[OFF_K + krow * QP + bk2];
                Bm.u[1] = *(const uint4*)&sh[OFF_K + krow * QP + bk2 + 16];
                acc = __builtin_amdgcn_wmma_f32_16x16x32_f16(false, A.h, false, Bm.h,
                                                             (short)0, acc, false, false);
            }
            const int kidx  = nt * 16 + ln;
            const int ktime = kidx / 22;
#pragma unroll
            for (int j = 0; j < 8; ++j) {
                const int qidx = q0 + j + hiK8;
                const bool keep = (qidx == kidx) || ((qidx / 22) != ktime);
                float v = keep ? acc[j] * 0.125f : NEGV;
                sc[nt][j] = v;
                mx[j] = fmaxf(mx[j], v);
            }
        }
        // row-max across the 16 N-lanes of each half-wave
#pragma unroll
        for (int j = 0; j < 8; ++j) {
#pragma unroll
            for (int off = 8; off >= 1; off >>= 1)
                mx[j] = fmaxf(mx[j], __shfl_xor(mx[j], off, 32));
        }
        float sm[8];
#pragma unroll
        for (int j = 0; j < 8; ++j) sm[j] = 0.f;
#pragma unroll
        for (int nt = 0; nt < 11; ++nt) {
#pragma unroll
            for (int j = 0; j < 8; ++j) {
                float e = __expf(sc[nt][j] - mx[j]);
                sm[j] += e;
                sh[pbase + (j + hiK8) * VP + nt * 16 + ln] = (_Float16)e;  // P row-major
            }
        }
#pragma unroll
        for (int j = 0; j < 8; ++j) {
#pragma unroll
            for (int off = 8; off >= 1; off >>= 1)
                sm[j] += __shfl_xor(sm[j], off, 32);
        }
        if (lane == 0 || lane == 16) {
#pragma unroll
            for (int j = 0; j < 8; ++j) sums[w * 16 + hiK8 + j] = sm[j];
        }
        const float invs = 1.0f / sums[w * 16 + ln];

        // out^T tile = V^T (16d x 192keys) * P^T (192keys x 16q), then scaled store
        float* orow = out + ((size_t)b * SEQ + (q0 + ln)) * DMODEL + h * HDIM;
#pragma unroll
        for (int dt = 0; dt < 4; ++dt) {
            v8f acc = {};
            const int vrow = dt * 16 + ln;
#pragma unroll
            for (int ks = 0; ks < 6; ++ks) {
                AB16 A, Bm;
                const int ak = ks * 32 + hiK8;
                A.u[0]  = *(const uint4*)&sh[OFF_VT + vrow * VP + ak];
                A.u[1]  = *(const uint4*)&sh[OFF_VT + vrow * VP + ak + 16];
                const int bk2 = ks * 32 + hiK8;
                Bm.u[0] = *(const uint4*)&sh[pbase + ln * VP + bk2];
                Bm.u[1] = *(const uint4*)&sh[pbase + ln * VP + bk2 + 16];
                acc = __builtin_amdgcn_wmma_f32_16x16x32_f16(false, A.h, false, Bm.h,
                                                             (short)0, acc, false, false);
            }
            float4 f0, f1;
            f0.x = acc[0] * invs; f0.y = acc[1] * invs;
            f0.z = acc[2] * invs; f0.w = acc[3] * invs;
            f1.x = acc[4] * invs; f1.y = acc[5] * invs;
            f1.z = acc[6] * invs; f1.w = acc[7] * invs;
            float* op = orow + dt * 16 + hiK8;
            *(float4*)op       = f0;
            *(float4*)(op + 4) = f1;
        }
    }
}

extern "C" void kernel_launch(void* const* d_in, const int* in_sizes, int n_in,
                              void* d_out, int out_size, void* d_ws, size_t ws_size,
                              hipStream_t stream) {
    (void)in_sizes; (void)n_in; (void)out_size; (void)d_ws; (void)ws_size;
    const float* x  = (const float*)d_in[0];
    const float* Wq = (const float*)d_in[1];
    const float* bq = (const float*)d_in[2];
    const float* Wk = (const float*)d_in[3];
    const float* bk = (const float*)d_in[4];
    const float* Wv = (const float*)d_in[5];
    const float* bv = (const float*)d_in[6];
    float* out = (float*)d_out;

    dim3 grid(512 * HNUM);   // one block per (batch, head)
    dim3 block(256);         // 8 waves (wave32)
    mha_temporal_kernel<<<grid, block, 0, stream>>>(x, Wq, bq, Wk, bk, Wv, bv, out);
}
